// GCN_38981123178606
// MI455X (gfx1250) — matrix operand order repaired and checked by reading, not compiled, not run on hardware
//
#include <hip/hip_runtime.h>

// ---- types ------------------------------------------------------------
typedef __attribute__((ext_vector_type(16))) __bf16 bf16x16;
typedef __attribute__((ext_vector_type(8)))  __bf16 bf16x8;
typedef __attribute__((ext_vector_type(8)))  float  f32x8;
typedef __attribute__((ext_vector_type(4)))  float  f32x4;
typedef __attribute__((ext_vector_type(4)))  unsigned int u32x4;
typedef __attribute__((ext_vector_type(8)))  int    i32x8;
typedef __attribute__((ext_vector_type(4)))  int    i32x4;

union FragBF16 { bf16x16 v; bf16x8 h[2]; };

#define LDS_STRIDE 40   // 32 bf16 payload + 8 pad = 16 DWORDs + 4 DWORDs pad (TDM-expressible)

#if defined(__AMDGCN__) && __has_builtin(__builtin_amdgcn_tensor_load_to_lds) && \
    __has_builtin(__builtin_amdgcn_s_wait_tensorcnt)
#define USE_TDM 1
#else
#define USE_TDM 0
#endif

// ---------------------------------------------------------------------
// Kernel 0: Wt[e][k] = bf16(W[k][e])   (256x256)
// ---------------------------------------------------------------------
__global__ void gcn_wt_kernel(const float* __restrict__ W, __bf16* __restrict__ Wt) {
    int idx = blockIdx.x * 256 + threadIdx.x;
    int k = idx >> 8;
    int e = idx & 255;
    Wt[e * 256 + k] = (__bf16)W[k * 256 + e];
}

// ---------------------------------------------------------------------
// Unified bf16 WMMA GEMM:  C[m][n] = sum_k A_f32[m][k] * Bt_bf16[n][k]
//   KTOT : K extent, LDAG : A row stride (f32), LDBG : Bt row stride (bf16)
//   EPI 0: write bf16 transposed into Ft[(b*256+e)*4096 + nseq]  (GEMM1)
//   EPI 1: bias + PReLU, write f32 out[(b*4096+m)*256 + e]       (GEMM2)
// Block tile 128(M) x 128(N), 256 threads = 8 waves (2 M-halves x 4 N-quarters),
// each wave computes 4x2 WMMA 16x16 tiles. B tile staged via Tensor Data Mover
// (TENSOR_LOAD_TO_LDS) with LDS row padding done by the TDM itself.
// ---------------------------------------------------------------------
template<int KTOT, int LDAG, int LDBG, int EPI>
__global__ __launch_bounds__(256) void gcn_gemm_kernel(
    const float*  __restrict__ A,
    const __bf16* __restrict__ Bt,
    __bf16*       __restrict__ FtOut,
    const float*  __restrict__ bias,
    const float*  __restrict__ prelu,
    float*        __restrict__ Out)
{
    __shared__ __bf16 Al[128 * LDS_STRIDE];
    __shared__ __bf16 Bl[128 * LDS_STRIDE];

    const int t    = threadIdx.x;
    const int lane = t & 31;
    const int wave = t >> 5;
    const int wm   = wave & 1;   // M half   (0..1) -> 64 rows each
    const int wn   = wave >> 1;  // N quarter(0..3) -> 32 cols each

    const int m0 = blockIdx.x * 128;
    const int n0 = blockIdx.y * 128;
    const int b  = blockIdx.z;

    const float*  Ab = A  + (size_t)b * 4096 * (size_t)LDAG;  // b==0 for GEMM1
    const __bf16* Bb = Bt + (size_t)b *  256 * (size_t)LDBG;  // b==0 for GEMM1

    f32x8 acc[4][2];
#pragma unroll
    for (int i = 0; i < 4; ++i)
#pragma unroll
        for (int j = 0; j < 2; ++j)
#pragma unroll
            for (int e = 0; e < 8; ++e) acc[i][j][e] = 0.0f;

    // per-thread staging coordinates
    const int mmA = t >> 1;            // A tile row (0..127)
    const int khA = (t & 1) * 16;      // A k half offset (0 / 16)

#if USE_TDM
    // ---- TDM descriptor (constant parts) -------------------------------
    // Tile: dim0 = 32 k-elements (64B rows), dim1 = 128 n-rows.
    // LDS padding: every 16 DWORDs (pad_interval=3) insert 4 DWORDs (pad_amount=3)
    // -> row pitch 80B = LDS_STRIDE bf16, matching the fragment addressing.
    const unsigned lds_base =
        (unsigned)(size_t)(__attribute__((address_space(3))) __bf16*)&Bl[0];
    i32x8 g1;
    g1[0] = (int)((1u << 16) |            // data_size = 1 (2 bytes)
                  (1u << 20) |            // pad_enable
                  (3u << 22) |            // pad_interval: 16 DWORDs
                  (3u << 25));            // pad_amount:   4 DWORDs
    g1[1] = (int)(32u << 16);             // tensor_dim0 = 32 (lo16 in [31:16])
    g1[2] = (int)(128u << 16);            // tensor_dim0 hi=0, tensor_dim1 = 128
    g1[3] = (int)(32u << 16);             // tensor_dim1 hi=0, tile_dim0 = 32
    g1[4] = (int)(128u);                  // tile_dim1 = 128, tile_dim2 = 0
    g1[5] = (int)((unsigned)LDBG);        // tensor_dim0_stride lo32
    g1[6] = 0;                            // stride hi, tensor_dim1_stride lo
    g1[7] = 0;
    const i32x4 gz  = {0, 0, 0, 0};       // groups 2/3 unused (2D tile)
    const i32x8 gz8 = {0, 0, 0, 0, 0, 0, 0, 0};
#else
    const int nnB = t >> 1;            // B tile row (n index, 0..127)
    const int kcB = (t & 1) * 16;      // B k chunk  (0 / 16)
#endif

    for (int k0 = 0; k0 < KTOT; k0 += 32) {
#if USE_TDM
        // ---- stage B via Tensor Data Mover (one DMA per k-step) ----
        if (wave == 0) {
            const unsigned long long ga =
                (unsigned long long)(const void*)(Bb + (size_t)n0 * LDBG + k0);
            u32x4 g0;
            g0[0] = 1u;                               // count=1, no gather
            g0[1] = lds_base;                         // lds_addr
            g0[2] = (unsigned)(ga & 0xFFFFFFFFull);   // global_addr lo
            g0[3] = (unsigned)((ga >> 32) & 0x1FFFFFFull) | (2u << 30); // hi + type=2
            __builtin_amdgcn_tensor_load_to_lds(g0, g1, gz, gz, gz8, 0);
        }
#else
        // ---- stage B: 128x32 bf16 straight copy (rows = n, contiguous k) ----
        {
            const __bf16* src = Bb + (size_t)(n0 + nnB) * LDBG + k0 + kcB;
            u32x4 q0 = *(const u32x4*)(src);
            u32x4 q1 = *(const u32x4*)(src + 8);
            *(u32x4*)&Bl[nnB * LDS_STRIDE + kcB]     = q0;
            *(u32x4*)&Bl[nnB * LDS_STRIDE + kcB + 8] = q1;
        }
#endif
        // ---- stage A: 128x32 f32 -> bf16 -> LDS ----
        {
            const float* src = Ab + (size_t)(m0 + mmA) * LDAG + k0 + khA;
            f32x4 f0 = *(const f32x4*)(src + 0);
            f32x4 f1 = *(const f32x4*)(src + 4);
            f32x4 f2 = *(const f32x4*)(src + 8);
            f32x4 f3 = *(const f32x4*)(src + 12);
            if (k0 + 32 < KTOT)
                __builtin_prefetch(src + 32, 0, 3);
            bf16x8 p0, p1;
#pragma unroll
            for (int e = 0; e < 4; ++e) {
                p0[e]     = (__bf16)f0[e];
                p0[4 + e] = (__bf16)f1[e];
                p1[e]     = (__bf16)f2[e];
                p1[4 + e] = (__bf16)f3[e];
            }
            *(bf16x8*)&Al[mmA * LDS_STRIDE + khA]     = p0;
            *(bf16x8*)&Al[mmA * LDS_STRIDE + khA + 8] = p1;
        }
#if USE_TDM
        if (wave == 0)
            __builtin_amdgcn_s_wait_tensorcnt(0);   // B tile fully in LDS
#endif
        __syncthreads();

        // ---- fragments (ISA bf16 16x32 A / 32x16 B layouts) ----
        FragBF16 afr[4];
#pragma unroll
        for (int mt = 0; mt < 4; ++mt) {
            const int am  = wm * 64 + mt * 16 + (lane & 15);
            const int akb = (lane >> 4) * 8;
            afr[mt].h[0] = *(const bf16x8*)&Al[am * LDS_STRIDE + akb];
            afr[mt].h[1] = *(const bf16x8*)&Al[am * LDS_STRIDE + akb + 16];
        }
        FragBF16 bfr[2];
#pragma unroll
        for (int nt = 0; nt < 2; ++nt) {
            const int bn  = wn * 32 + nt * 16 + (lane & 15);
            const int bkb = (lane >> 4) * 8;
            bfr[nt].h[0] = *(const bf16x8*)&Bl[bn * LDS_STRIDE + bkb];
            bfr[nt].h[1] = *(const bf16x8*)&Bl[bn * LDS_STRIDE + bkb + 16];
        }

#pragma unroll
        for (int mt = 0; mt < 4; ++mt)
#pragma unroll
            for (int nt = 0; nt < 2; ++nt)
                acc[mt][nt] = __builtin_amdgcn_wmma_f32_16x16x32_bf16(
                    false, afr[mt].v, false, bfr[nt].v,
                    (short)0, acc[mt][nt], false, false);

        __syncthreads();   // all waves done reading LDS before next stage/TDM
    }

    // ---- epilogue ----
    if constexpr (EPI == 0) {
        // write bf16, transposed: Ft[(b'*256 + e)*4096 + nseq], m -> (b', nseq)
#pragma unroll
        for (int mt = 0; mt < 4; ++mt)
#pragma unroll
            for (int nt = 0; nt < 2; ++nt) {
                const int e     = n0 + wn * 32 + nt * 16 + (lane & 15);
                const int mbase = m0 + wm * 64 + mt * 16 + (lane >> 4) * 8;
#pragma unroll
                for (int r = 0; r < 8; ++r) {
                    const int m  = mbase + r;
                    const int bb = m >> 12;       // batch
                    const int ns = m & 4095;      // sequence index
                    FtOut[((size_t)bb * 256 + e) * 4096 + ns] = (__bf16)acc[mt][nt][r];
                }
            }
    } else {
        const float a = *prelu;
#pragma unroll
        for (int mt = 0; mt < 4; ++mt)
#pragma unroll
            for (int nt = 0; nt < 2; ++nt) {
                const int e     = n0 + wn * 32 + nt * 16 + (lane & 15);
                const float bv  = bias[e];
                const int mbase = m0 + wm * 64 + mt * 16 + (lane >> 4) * 8;
#pragma unroll
                for (int r = 0; r < 8; ++r) {
                    float v = acc[mt][nt][r] + bv;
                    v = (v >= 0.0f) ? v : a * v;
                    Out[((size_t)b * 4096 + mbase + r) * 256 + e] = v;
                }
            }
    }
}

// ---------------------------------------------------------------------
extern "C" void kernel_launch(void* const* d_in, const int* in_sizes, int n_in,
                              void* d_out, int out_size, void* d_ws, size_t ws_size,
                              hipStream_t stream) {
    const float* seq  = (const float*)d_in[0];  // [4,4096,256]
    const float* adj  = (const float*)d_in[1];  // [4,4096,4096]
    const float* W    = (const float*)d_in[2];  // [256,256]
    const float* bias = (const float*)d_in[3];  // [256]
    const float* pa   = (const float*)d_in[4];  // scalar
    float* out        = (float*)d_out;          // [4,4096,256]

    char* ws = (char*)d_ws;
    __bf16* Wt = (__bf16*)ws;                    // 256*256*2   = 128 KB
    __bf16* Ft = (__bf16*)(ws + (256u << 10));   // 4*256*4096*2 = 8 MB

    // W -> Wt (transpose + bf16)
    gcn_wt_kernel<<<256, 256, 0, stream>>>(W, Wt);

    // GEMM1: Ft[e][nseq] (bf16) = seq @ W ;  M=16384 (batch folded), K=256
    gcn_gemm_kernel<256, 256, 256, 0>
        <<<dim3(128, 2, 1), 256, 0, stream>>>(seq, Wt, Ft, nullptr, nullptr, nullptr);

    // GEMM2: out = PReLU(adj @ seq_fts + bias) ;  per batch M=4096, K=4096
    gcn_gemm_kernel<4096, 4096, 4096, 1>
        <<<dim3(32, 2, 4), 256, 0, stream>>>(adj, Ft, nullptr, bias, pa, out);
}
